// DenseOrderOneTerm_34170759807425
// MI455X (gfx1250) — compile-verified, matrix operand order checked by reading, not run on hardware
//
#include <hip/hip_runtime.h>
#include <hip/hip_bf16.h>

// ---------------------------------------------------------------------------
// Fused TimeInjectionNet MLP + batched 32x32 expm for MI455X (gfx1250).
// bf16 WMMA (v_wmma_f32_16x16x32_bf16) everywhere; weights converted and
// transposed once into workspace so B-fragments are contiguous per-lane loads.
// MT=32 rows per workgroup: halves L2 weight traffic vs MT=16 and reuses each
// weight fragment across two M-tiles.
// ---------------------------------------------------------------------------

typedef __bf16 bf16_t;
typedef __attribute__((ext_vector_type(16))) __bf16 v16bf;
typedef __attribute__((ext_vector_type(8)))  float  v8f;

#define B_TOT   16384
#define D_DIM   32
#define H_DIM   512
#define DD_DIM  1024
#define MT      32      // batch rows per workgroup
#define XPITCH  520     // bf16 elems: 1040B rows -> 16B aligned, bank-spread
#define DPITCH  1040
#define EPITCH  40      // expm scratch pitch (80B rows, 16B aligned)
#define N_TAYLOR 12
#define N_SQUARE 5

__device__ __forceinline__ bf16_t f2bf(float f) {
  unsigned u = __builtin_bit_cast(unsigned, f);
  unsigned r = u + 0x7FFFu + ((u >> 16) & 1u);      // round-to-nearest-even
  unsigned short h = (unsigned short)(r >> 16);
  return __builtin_bit_cast(bf16_t, h);
}
__device__ __forceinline__ float bf2f(bf16_t b) {
  unsigned short h = __builtin_bit_cast(unsigned short, b);
  unsigned u = ((unsigned)h) << 16;
  return __builtin_bit_cast(float, u);
}
__device__ __forceinline__ float selu_f(float v) {
  const float a = 1.6732632423543772f, s = 1.0507009873554805f;
  return s * (v > 0.f ? v : a * (__expf(v) - 1.f));
}
__device__ __forceinline__ v8f wmma_bf16(v16bf a, v16bf b, v8f c) {
  return __builtin_amdgcn_wmma_f32_16x16x32_bf16(false, a, false, b,
                                                 (short)0, c, false, false);
}

// ------------------------- weight prep: f32 -> bf16, transposed -------------
// W is [(K+1) x N] row-major (last row = time row). Wt is [N x K] bf16.
__global__ __launch_bounds__(256) void cvt_weights(const float* __restrict__ W,
                                                   bf16_t* __restrict__ Wt,
                                                   float* __restrict__ trow,
                                                   int K, int N) {
  int idx = blockIdx.x * blockDim.x + threadIdx.x;
  if (idx < K * N) {
    int n = idx / K, k = idx % K;
    Wt[idx] = f2bf(W[(size_t)k * N + n]);
  }
  if (idx < N) trow[idx] = W[(size_t)K * N + idx];
}

// ------------------------------ fused kernel --------------------------------
__global__ __launch_bounds__(256) void fused_mlp_expm(
    const float* __restrict__ q,  const float* __restrict__ p,
    const float* __restrict__ t,  const float* __restrict__ dtp,
    const float* __restrict__ b0, const float* __restrict__ b1,
    const float* __restrict__ b2, const float* __restrict__ b3,
    const float* __restrict__ bout,
    const bf16_t* __restrict__ wt0, const bf16_t* __restrict__ wt1,
    const bf16_t* __restrict__ wt2, const bf16_t* __restrict__ wt3,
    const bf16_t* __restrict__ wtout,
    const float* __restrict__ tr0, const float* __restrict__ tr1,
    const float* __restrict__ tr2, const float* __restrict__ tr3,
    const float* __restrict__ trout,
    float* __restrict__ out_q, float* __restrict__ out_dlogp) {

  __shared__ bf16_t sX[MT][XPITCH];          // activations (bf16)
  __shared__ bf16_t sD[MT][DPITCH];          // dense 32x32 per row (bf16, clipped)
  __shared__ float  sQ[MT][D_DIM];
  __shared__ float  sT[MT];
  __shared__ bf16_t sE[8][2][32][EPITCH];    // per-wave expm scratch: [0]=T, [1]=T^t

  const int tid  = threadIdx.x;
  const int wave = tid >> 5;
  const int lane = tid & 31;
  const int lhalf = lane >> 4;               // K-half owned by this lane
  const int lcol  = lane & 15;               // M (A-frag) / N (B,C-frag) index
  const int base = blockIdx.x * MT;
  const float dt = dtp[0];

  // ---- stage p (bf16), q, t into LDS ----
  for (int i = tid; i < MT * D_DIM; i += 256) {
    int m = i >> 5, k = i & 31;
    sX[m][k] = f2bf(p[(size_t)(base + m) * D_DIM + k]);
    sQ[m][k] = q[(size_t)(base + m) * D_DIM + k];
  }
  if (tid < MT) sT[tid] = t[base + tid];
  __syncthreads();

  // ================= layer 0: [32x32] @ [32x512]  (K = 32) =================
  {
    v16bf a[2];
#pragma unroll
    for (int mt = 0; mt < 2; ++mt)
      a[mt] = *(const v16bf*)&sX[mt * 16 + lcol][lhalf * 16];
    v8f c[2][4];
#pragma unroll
    for (int nt = 0; nt < 4; ++nt) {
      int ncol = wave * 64 + nt * 16 + lcol;
      v16bf bf = *(const v16bf*)&wt0[(size_t)ncol * D_DIM + lhalf * 16];
#pragma unroll
      for (int mt = 0; mt < 2; ++mt) {
        v8f z = {};
        c[mt][nt] = wmma_bf16(a[mt], bf, z);
      }
    }
    __syncthreads();
#pragma unroll
    for (int nt = 0; nt < 4; ++nt) {
      int ncol = wave * 64 + nt * 16 + lcol;
      float trn = tr0[ncol], bn = b0[ncol];
#pragma unroll
      for (int mt = 0; mt < 2; ++mt)
#pragma unroll
        for (int j = 0; j < 8; ++j) {
          int m = mt * 16 + lhalf * 8 + j;
          sX[m][ncol] = f2bf(selu_f(c[mt][nt][j] + sT[m] * trn + bn));
        }
    }
    __syncthreads();
  }

  // ================= hidden layers 1..3: [32x512] @ [512x512] ==============
  const bf16_t* wts[3] = {wt1, wt2, wt3};
  const float*  trs[3] = {tr1, tr2, tr3};
  const float*  bss[3] = {b1, b2, b3};
  for (int L = 0; L < 3; ++L) {
    const bf16_t* wl = wts[L];
    v8f c[2][4];
#pragma unroll
    for (int mt = 0; mt < 2; ++mt)
#pragma unroll
      for (int nt = 0; nt < 4; ++nt) { v8f z = {}; c[mt][nt] = z; }
#pragma unroll
    for (int kk = 0; kk < H_DIM; kk += 32) {
      v16bf a[2];
#pragma unroll
      for (int mt = 0; mt < 2; ++mt)
        a[mt] = *(const v16bf*)&sX[mt * 16 + lcol][kk + lhalf * 16];
      if (kk + 32 < H_DIM)
        __builtin_prefetch(&wl[(size_t)(wave * 64 + lcol) * H_DIM + kk + 32], 0, 1);
#pragma unroll
      for (int nt = 0; nt < 4; ++nt) {
        int ncol = wave * 64 + nt * 16 + lcol;
        v16bf bf = *(const v16bf*)&wl[(size_t)ncol * H_DIM + kk + lhalf * 16];
#pragma unroll
        for (int mt = 0; mt < 2; ++mt)
          c[mt][nt] = wmma_bf16(a[mt], bf, c[mt][nt]);
      }
    }
    __syncthreads();
    const float* trl = trs[L];
    const float* bl  = bss[L];
#pragma unroll
    for (int nt = 0; nt < 4; ++nt) {
      int ncol = wave * 64 + nt * 16 + lcol;
      float trn = trl[ncol], bn = bl[ncol];
#pragma unroll
      for (int mt = 0; mt < 2; ++mt)
#pragma unroll
        for (int j = 0; j < 8; ++j) {
          int m = mt * 16 + lhalf * 8 + j;
          sX[m][ncol] = f2bf(selu_f(c[mt][nt][j] + sT[m] * trn + bn));
        }
    }
    __syncthreads();
  }

  // ============== output layer: [32x512] @ [512x1024] -> dense =============
  {
    v8f c[2][8];
#pragma unroll
    for (int mt = 0; mt < 2; ++mt)
#pragma unroll
      for (int nt = 0; nt < 8; ++nt) { v8f z = {}; c[mt][nt] = z; }
#pragma unroll
    for (int kk = 0; kk < H_DIM; kk += 32) {
      v16bf a[2];
#pragma unroll
      for (int mt = 0; mt < 2; ++mt)
        a[mt] = *(const v16bf*)&sX[mt * 16 + lcol][kk + lhalf * 16];
#pragma unroll
      for (int nt = 0; nt < 8; ++nt) {
        int ncol = wave * 128 + nt * 16 + lcol;
        v16bf bf = *(const v16bf*)&wtout[(size_t)ncol * H_DIM + kk + lhalf * 16];
#pragma unroll
        for (int mt = 0; mt < 2; ++mt)
          c[mt][nt] = wmma_bf16(a[mt], bf, c[mt][nt]);
      }
    }
    __syncthreads();
#pragma unroll
    for (int nt = 0; nt < 8; ++nt) {
      int ncol = wave * 128 + nt * 16 + lcol;
      float trn = trout[ncol], bn = bout[ncol];
#pragma unroll
      for (int mt = 0; mt < 2; ++mt)
#pragma unroll
        for (int j = 0; j < 8; ++j) {
          int m = mt * 16 + lhalf * 8 + j;
          float v = c[mt][nt][j] + sT[m] * trn + bn;
          v = fminf(20.f, fmaxf(-20.f, v));         // clip
          sD[m][ncol] = f2bf(v);
        }
    }
    __syncthreads();
  }

  // =========== expm per batch row (4 matrices per wave, private) ============
  bf16_t (*Tb)[EPITCH]  = sE[wave][0];
  bf16_t (*TbT)[EPITCH] = sE[wave][1];
  const float sc = dt * (1.0f / 32.0f);             // 2^-5 scaling

  for (int mi = 0; mi < 4; ++mi) {
    const int r = wave * 4 + mi;                    // local batch row

    // --- trace -> dlogp (dense already clipped) ---
    float dv = bf2f(sD[r][lane * 33]);              // diag elem i = lane
    for (int off = 16; off; off >>= 1) dv += __shfl_down(dv, off);
    if (lane == 0) out_dlogp[base + r] = dt * dv;

    // --- A fragments: A = (dt/32) * dense, rows 0-15 and 16-31 ---
    v16bf afr[2];
#pragma unroll
    for (int h = 0; h < 2; ++h) {
      v16bf raw = *(const v16bf*)&sD[r][(h * 16 + lcol) * 32 + lhalf * 16];
      v16bf cv;
#pragma unroll
      for (int e = 0; e < 16; ++e) cv[e] = f2bf(bf2f(raw[e]) * sc);
      afr[h] = cv;
    }

    // --- T = I (both orientations) ---
#pragma unroll 1
    for (int j = 0; j < 32; ++j) {
      bf16_t v = f2bf(j == lane ? 1.f : 0.f);
      Tb[lane][j] = v;
      TbT[lane][j] = v;
    }

    // --- Horner Taylor: T <- I + (A @ T) / k ---
#pragma unroll 1
    for (int term = N_TAYLOR; term >= 1; --term) {
      const float inv = 1.0f / (float)term;
      v8f c[4];
#pragma unroll
      for (int i = 0; i < 4; ++i) { v8f z = {}; c[i] = z; }
#pragma unroll
      for (int mh = 0; mh < 2; ++mh)
#pragma unroll
        for (int nh = 0; nh < 2; ++nh) {
          v16bf bf = *(const v16bf*)&TbT[nh * 16 + lcol][lhalf * 16];
          c[mh * 2 + nh] = wmma_bf16(afr[mh], bf, c[mh * 2 + nh]);
        }
#pragma unroll
      for (int mh = 0; mh < 2; ++mh)
#pragma unroll
        for (int nh = 0; nh < 2; ++nh)
#pragma unroll
          for (int j = 0; j < 8; ++j) {
            int m = mh * 16 + lhalf * 8 + j;
            int n = nh * 16 + lcol;
            float v = c[mh * 2 + nh][j] * inv + (m == n ? 1.f : 0.f);
            bf16_t bv = f2bf(v);
            Tb[m][n] = bv;
            TbT[n][m] = bv;
          }
    }

    // --- s squarings: T <- T @ T ---
#pragma unroll 1
    for (int sq = 0; sq < N_SQUARE; ++sq) {
      v16bf ta[2];
#pragma unroll
      for (int h = 0; h < 2; ++h)
        ta[h] = *(const v16bf*)&Tb[h * 16 + lcol][lhalf * 16];
      v8f c[4];
#pragma unroll
      for (int i = 0; i < 4; ++i) { v8f z = {}; c[i] = z; }
#pragma unroll
      for (int mh = 0; mh < 2; ++mh)
#pragma unroll
        for (int nh = 0; nh < 2; ++nh) {
          v16bf bf = *(const v16bf*)&TbT[nh * 16 + lcol][lhalf * 16];
          c[mh * 2 + nh] = wmma_bf16(ta[mh], bf, c[mh * 2 + nh]);
        }
#pragma unroll
      for (int mh = 0; mh < 2; ++mh)
#pragma unroll
        for (int nh = 0; nh < 2; ++nh)
#pragma unroll
          for (int j = 0; j < 8; ++j) {
            int m = mh * 16 + lhalf * 8 + j;
            int n = nh * 16 + lcol;
            bf16_t bv = f2bf(c[mh * 2 + nh][j]);
            Tb[m][n] = bv;
            TbT[n][m] = bv;
          }
    }

    // --- new_q = expm @ q : lane = output row ---
    float acc = 0.f;
#pragma unroll 1
    for (int n = 0; n < 32; ++n) acc += bf2f(Tb[lane][n]) * sQ[r][n];
    out_q[(size_t)(base + r) * D_DIM + lane] = acc;
  }
}

// ------------------------------- launcher -----------------------------------
extern "C" void kernel_launch(void* const* d_in, const int* in_sizes, int n_in,
                              void* d_out, int out_size, void* d_ws, size_t ws_size,
                              hipStream_t stream) {
  const float* q    = (const float*)d_in[0];
  const float* p    = (const float*)d_in[1];
  const float* t    = (const float*)d_in[2];
  const float* dt   = (const float*)d_in[3];
  const float* W0   = (const float*)d_in[4];
  const float* b0   = (const float*)d_in[5];
  const float* W1   = (const float*)d_in[6];
  const float* b1   = (const float*)d_in[7];
  const float* W2   = (const float*)d_in[8];
  const float* b2   = (const float*)d_in[9];
  const float* W3   = (const float*)d_in[10];
  const float* b3   = (const float*)d_in[11];
  const float* Wout = (const float*)d_in[12];
  const float* bout = (const float*)d_in[13];

  char* ws = (char*)d_ws;
  size_t o = 0;
  bf16_t* wt0   = (bf16_t*)(ws + o); o += (size_t)H_DIM * D_DIM * 2;
  bf16_t* wt1   = (bf16_t*)(ws + o); o += (size_t)H_DIM * H_DIM * 2;
  bf16_t* wt2   = (bf16_t*)(ws + o); o += (size_t)H_DIM * H_DIM * 2;
  bf16_t* wt3   = (bf16_t*)(ws + o); o += (size_t)H_DIM * H_DIM * 2;
  bf16_t* wtout = (bf16_t*)(ws + o); o += (size_t)DD_DIM * H_DIM * 2;
  float* tr0    = (float*)(ws + o);  o += H_DIM * 4;
  float* tr1    = (float*)(ws + o);  o += H_DIM * 4;
  float* tr2    = (float*)(ws + o);  o += H_DIM * 4;
  float* tr3    = (float*)(ws + o);  o += H_DIM * 4;
  float* trout  = (float*)(ws + o);  o += DD_DIM * 4;

  // weight prep (transpose + bf16 convert)
  cvt_weights<<<(D_DIM * H_DIM + 255) / 256, 256, 0, stream>>>(W0, wt0, tr0, D_DIM, H_DIM);
  cvt_weights<<<(H_DIM * H_DIM + 255) / 256, 256, 0, stream>>>(W1, wt1, tr1, H_DIM, H_DIM);
  cvt_weights<<<(H_DIM * H_DIM + 255) / 256, 256, 0, stream>>>(W2, wt2, tr2, H_DIM, H_DIM);
  cvt_weights<<<(H_DIM * H_DIM + 255) / 256, 256, 0, stream>>>(W3, wt3, tr3, H_DIM, H_DIM);
  cvt_weights<<<(H_DIM * DD_DIM + 255) / 256, 256, 0, stream>>>(Wout, wtout, trout, H_DIM, DD_DIM);

  float* out_q     = (float*)d_out;
  float* out_dlogp = out_q + (size_t)B_TOT * D_DIM;

  fused_mlp_expm<<<B_TOT / MT, 256, 0, stream>>>(
      q, p, t, dt, b0, b1, b2, b3, bout,
      wt0, wt1, wt2, wt3, wtout,
      tr0, tr1, tr2, tr3, trout,
      out_q, out_dlogp);
}
